// TernaryLinear_12455405158913
// MI455X (gfx1250) — matrix operand order repaired and checked by reading, not compiled, not run on hardware
//
#include <hip/hip_runtime.h>

// CDNA5 (gfx1250) wave32 WMMA types
typedef __attribute__((ext_vector_type(16))) _Float16 v16h;
typedef __attribute__((ext_vector_type(8)))  float    v8f;
typedef __attribute__((ext_vector_type(4)))  float    vf4;
typedef __attribute__((ext_vector_type(4)))  _Float16 v4h;

#define RED_BLOCKS  1024
#define RED_THREADS 256
#define TM 128
#define TN 128
#define TK 32

#if defined(__gfx1250__) && __has_builtin(__builtin_amdgcn_global_load_async_to_lds_b128)
#define ASYNC_B 1
#else
#define ASYNC_B 0
#endif

// 8x f32 -> 8x f16 packed (4x v_cvt_pk_f16_f32)
__device__ __forceinline__ uint4 cvt8_f16(vf4 a, vf4 b) {
    union { v4h h[2]; uint4 q; } u;
    u.h[0] = __builtin_convertvector(a, v4h);
    u.h[1] = __builtin_convertvector(b, v4h);
    return u.q;
}

#if ASYNC_B
// builtin signature (from probe diagnostics):
//   arg0 = AS(1) int4*  (global source), arg1 = AS(3) int4* (LDS dest),
//   arg2 = imm offset, arg3 = cpol
typedef __attribute__((__vector_size__(4 * sizeof(int)))) int gv4i;
typedef __attribute__((address_space(1))) gv4i* gptr_t;
typedef __attribute__((address_space(3))) gv4i* lptr_t;

__device__ __forceinline__ gptr_t gaddr(const void* p) {
    return (gptr_t)(unsigned long long)(uintptr_t)p;   // generic->AS(1), same bits
}
__device__ __forceinline__ lptr_t lds_addr(unsigned short* p) {
    // generic shared pointer low 32 bits == LDS offset
    return (lptr_t)(unsigned long long)(unsigned)(uintptr_t)p;
}
#endif

// ---------------- Pass 1a: partial sums of |w| (deterministic fixed partition) ----
__global__ __launch_bounds__(RED_THREADS)
void absmean_partial(const float* __restrict__ w, int nElem, double* __restrict__ partials) {
    const float4* w4 = (const float4*)w;               // nElem divisible by 4
    int n4 = nElem >> 2;
    int tid = blockIdx.x * blockDim.x + threadIdx.x;
    int stride = gridDim.x * blockDim.x;
    double s = 0.0;
    for (int i = tid; i < n4; i += stride) {
        float4 v = w4[i];
        s += (double)fabsf(v.x) + (double)fabsf(v.y) +
             (double)fabsf(v.z) + (double)fabsf(v.w);
    }
    __shared__ double red[RED_THREADS];
    red[threadIdx.x] = s;
    __syncthreads();
    for (int off = RED_THREADS / 2; off > 0; off >>= 1) {
        if (threadIdx.x < off) red[threadIdx.x] += red[threadIdx.x + off];
        __syncthreads();
    }
    if (threadIdx.x == 0) partials[blockIdx.x] = red[0];
}

// ---------------- Pass 1b: threshold = 0.7 * mean(|w|) ----
__global__ __launch_bounds__(RED_THREADS)
void finalize_threshold(const double* __restrict__ partials, int nPart, int nElem,
                        float* __restrict__ thr) {
    __shared__ double red[RED_THREADS];
    double s = 0.0;
    for (int i = threadIdx.x; i < nPart; i += RED_THREADS) s += partials[i];
    red[threadIdx.x] = s;
    __syncthreads();
    for (int off = RED_THREADS / 2; off > 0; off >>= 1) {
        if (threadIdx.x < off) red[threadIdx.x] += red[threadIdx.x + off];
        __syncthreads();
    }
    if (threadIdx.x == 0) *thr = 0.7f * (float)(red[0] / (double)nElem);
}

// ---------------- Pass 2: ternary-quantize weight -> fp16 {-1,0,+1} (exact) ----
__global__ __launch_bounds__(256)
void quantize_w(const float* __restrict__ w, const float* __restrict__ thrp,
                unsigned short* __restrict__ wq, int nElem) {
    const float thr = *thrp;
    int i = (blockIdx.x * blockDim.x + threadIdx.x) * 4;
    if (i >= nElem) return;
    float4 v = *(const float4*)(w + i);
    ushort4 q;
    q.x = (fabsf(v.x) >= thr) ? (v.x < 0.f ? 0xBC00u : 0x3C00u) : 0u;
    q.y = (fabsf(v.y) >= thr) ? (v.y < 0.f ? 0xBC00u : 0x3C00u) : 0u;
    q.z = (fabsf(v.z) >= thr) ? (v.z < 0.f ? 0xBC00u : 0x3C00u) : 0u;
    q.w = (fabsf(v.w) >= thr) ? (v.w < 0.f ? 0xBC00u : 0x3C00u) : 0u;
    *(ushort4*)(wq + i) = q;
}

// ---------------- Pass 3: WMMA GEMM  out = x @ Wq^T * scale + bias ----
// 256 threads = 8 waves (4 along M x 2 along N); block tile 128x128; K step 32.
__global__ __launch_bounds__(256)
void ternary_gemm(const float* __restrict__ X, const unsigned short* __restrict__ Wq,
                  const float* __restrict__ scale, const float* __restrict__ bias,
                  float* __restrict__ Out, int M, int N, int K) {
    __shared__ __align__(16) unsigned short lA[2][TM * TK];  // [m][k] fp16, 8 KB x2
    __shared__ __align__(16) unsigned short lB[2][TN * TK];  // [n][k] fp16, 8 KB x2

    const int tid  = threadIdx.x;
    const int lane = tid & 31;
    const int wave = tid >> 5;
    const int wm   = wave & 3;     // 0..3 : 32-row band
    const int wn   = wave >> 2;    // 0..1 : 64-col band
    const int m0   = blockIdx.y * TM;
    const int n0   = blockIdx.x * TN;

    // Staging map: 2 threads per row, 16 elements each
    const int sr = tid >> 1;
    const int sc = (tid & 1) * 16;
    const float*          Aptr = X  + (size_t)(m0 + sr) * K + sc;
    const unsigned short* Bptr = Wq + (size_t)(n0 + sr) * K + sc;

    union FragH { v16h v; uint4 q[2]; };
    const v8f vzero = {0.f, 0.f, 0.f, 0.f, 0.f, 0.f, 0.f, 0.f};
    v8f acc[2][4];
#pragma unroll
    for (int i = 0; i < 2; ++i)
#pragma unroll
        for (int j = 0; j < 4; ++j) acc[i][j] = vzero;

    // ---- prologue: fill LDS buffer 0 ----
    {
        vf4 a0 = *(const vf4*)(Aptr + 0);
        vf4 a1 = *(const vf4*)(Aptr + 4);
        vf4 a2 = *(const vf4*)(Aptr + 8);
        vf4 a3 = *(const vf4*)(Aptr + 12);
        uint4* as = (uint4*)&lA[0][sr * TK + sc];
        as[0] = cvt8_f16(a0, a1);
        as[1] = cvt8_f16(a2, a3);
#if ASYNC_B
        __builtin_amdgcn_global_load_async_to_lds_b128(
            gaddr(Bptr), lds_addr(&lB[0][sr * TK + sc]), 0, 0);
        __builtin_amdgcn_global_load_async_to_lds_b128(
            gaddr(Bptr + 8), lds_addr(&lB[0][sr * TK + sc + 8]), 0, 0);
        __asm volatile("s_wait_asynccnt 0x0" ::: "memory");
#else
        uint4 b0 = *(const uint4*)(Bptr + 0);
        uint4 b1 = *(const uint4*)(Bptr + 8);
        uint4* bs = (uint4*)&lB[0][sr * TK + sc];
        bs[0] = b0; bs[1] = b1;
#endif
    }
    __syncthreads();

    const int kTiles = K / TK;
    int buf = 0;
    for (int kt = 0; kt < kTiles; ++kt) {
        const bool hasNext = (kt + 1 < kTiles);

        // ---- kick off next tile's B copy (async, overlaps with WMMA) and A loads ----
        vf4 a0, a1, a2, a3;
#if !ASYNC_B
        uint4 b0, b1;
#endif
        if (hasNext) {
            const unsigned short* bp = Bptr + (size_t)(kt + 1) * TK;
#if ASYNC_B
            __builtin_amdgcn_global_load_async_to_lds_b128(
                gaddr(bp), lds_addr(&lB[buf ^ 1][sr * TK + sc]), 0, 0);
            __builtin_amdgcn_global_load_async_to_lds_b128(
                gaddr(bp + 8), lds_addr(&lB[buf ^ 1][sr * TK + sc + 8]), 0, 0);
#else
            b0 = *(const uint4*)(bp + 0);
            b1 = *(const uint4*)(bp + 8);
#endif
            const float* ap = Aptr + (size_t)(kt + 1) * TK;
            a0 = *(const vf4*)(ap + 0);
            a1 = *(const vf4*)(ap + 4);
            a2 = *(const vf4*)(ap + 8);
            a3 = *(const vf4*)(ap + 12);
        }

        // ---- load fragments from LDS (CDNA5 ISA VGPR layouts) ----
        const int r16  = lane & 15;
        const int half = lane >> 4;
        FragH afrag[2], bfrag[4];
#pragma unroll
        for (int i = 0; i < 2; ++i) {
            int row = wm * 32 + i * 16 + r16;
            // A 16-bit 16x32: VGPR0-3 hold K = half*8..+7, VGPR4-7 hold K = 16+half*8..+7
            afrag[i].q[0] = *(const uint4*)&lA[buf][row * TK + half * 8];
            afrag[i].q[1] = *(const uint4*)&lA[buf][row * TK + 16 + half * 8];
        }
#pragma unroll
        for (int j = 0; j < 4; ++j) {
            int col = wn * 64 + j * 16 + r16;
            // B 32x16: VGPR0-3 hold K = half*8..+7, VGPR4-7 hold K = 16+half*8..+7 (col n)
            bfrag[j].q[0] = *(const uint4*)&lB[buf][col * TK + half * 8];
            bfrag[j].q[1] = *(const uint4*)&lB[buf][col * TK + 16 + half * 8];
        }

        // ---- 8x v_wmma_f32_16x16x32_f16 ----
#pragma unroll
        for (int i = 0; i < 2; ++i)
#pragma unroll
            for (int j = 0; j < 4; ++j)
                acc[i][j] = __builtin_amdgcn_wmma_f32_16x16x32_f16(
                    false, afrag[i].v, false, bfrag[j].v,
                    (short)0, acc[i][j], false, false);

        // ---- stage next A tile (convert f32->f16) into the other LDS buffer ----
        if (hasNext) {
            uint4* as = (uint4*)&lA[buf ^ 1][sr * TK + sc];
            as[0] = cvt8_f16(a0, a1);
            as[1] = cvt8_f16(a2, a3);
#if !ASYNC_B
            uint4* bs = (uint4*)&lB[buf ^ 1][sr * TK + sc];
            bs[0] = b0; bs[1] = b1;
#endif
        }
#if ASYNC_B
        __asm volatile("s_wait_asynccnt 0x0" ::: "memory");
#endif
        __syncthreads();
        buf ^= 1;
    }

    // ---- epilogue: scale/bias in fp32, write out ----
    const int cn    = lane & 15;
    const int chalf = lane >> 4;
#pragma unroll
    for (int j = 0; j < 4; ++j) {
        int col   = n0 + wn * 64 + j * 16 + cn;
        float sc_ = scale[col];
        float bs_ = bias[col];
#pragma unroll
        for (int i = 0; i < 2; ++i) {
            int rbase = m0 + wm * 32 + i * 16 + chalf * 8;  // C: lanes16-31 -> M+8
#pragma unroll
            for (int v = 0; v < 8; ++v)
                Out[(size_t)(rbase + v) * N + col] = acc[i][j][v] * sc_ + bs_;
        }
    }
}

extern "C" void kernel_launch(void* const* d_in, const int* in_sizes, int n_in,
                              void* d_out, int out_size, void* d_ws, size_t ws_size,
                              hipStream_t stream) {
    const float* x     = (const float*)d_in[0];
    const float* w     = (const float*)d_in[1];
    const float* scale = (const float*)d_in[2];
    const float* bias  = (const float*)d_in[3];
    float* out = (float*)d_out;

    const int wsz = in_sizes[1];     // D_out * D_in
    const int N   = in_sizes[2];     // D_out (scale length)
    const int K   = wsz / N;         // D_in
    const int M   = in_sizes[0] / K; // B * S

    // workspace layout: [fp16 weight | double partials | float thr]
    unsigned short* wq = (unsigned short*)d_ws;
    size_t off = ((size_t)wsz * sizeof(unsigned short) + 255) & ~(size_t)255;
    double* partials = (double*)((char*)d_ws + off);
    float*  thr      = (float*)((char*)d_ws + off + RED_BLOCKS * sizeof(double));

    absmean_partial<<<RED_BLOCKS, RED_THREADS, 0, stream>>>(w, wsz, partials);
    finalize_threshold<<<1, RED_THREADS, 0, stream>>>(partials, RED_BLOCKS, wsz, thr);
    quantize_w<<<(wsz / 4 + 255) / 256, 256, 0, stream>>>(w, thr, wq, wsz);

    dim3 grid(N / TN, M / TM);   // shapes are multiples of 128 for this problem
    ternary_gemm<<<grid, 256, 0, stream>>>(x, wq, scale, bias, out, M, N, K);
}
